// FlashFFNMoE_48558900249352
// MI455X (gfx1250) — matrix operand order repaired and checked by reading, not compiled, not run on hardware
//
#include <hip/hip_runtime.h>

// Problem constants (match reference).
constexpr int Bc = 2, Hc = 8, Nc = 2048, Mc = 2048, Dc = 128;
constexpr int BH = Bc * Hc;     // 16 heads
constexpr int BQ = 128;         // query rows per workgroup (8 waves x 16 rows)
constexpr int TM = 32;          // keys per main-loop iteration
constexpr int NT = Mc / TM;     // 64 m-tiles

// LDS strides (elements), padded to avoid bank conflicts.
constexpr int SK = 136;  // bf16 stride for K/U tiles [32][128]
constexpr int SV = 40;   // bf16 stride for V^T tile  [128][32]
constexpr int SR = 36;   // f32  stride for R tile    [128][32] (TDM pad reproduces this)

typedef __attribute__((ext_vector_type(16))) __bf16    v16bf;
typedef __attribute__((ext_vector_type(8)))  float     v8f;
typedef __attribute__((ext_vector_type(8)))  unsigned  u32x8;
typedef __attribute__((ext_vector_type(4)))  unsigned  u32x4;
typedef __attribute__((ext_vector_type(8)))  int       i32x8;
typedef __attribute__((ext_vector_type(4)))  int       i32x4;

#if defined(__has_builtin)
#if __has_builtin(__builtin_amdgcn_tensor_load_to_lds) && \
    __has_builtin(__builtin_amdgcn_s_wait_tensorcnt)
#define HAVE_TDM 1
#endif
#if __has_builtin(__builtin_amdgcn_cvt_pk_bf16_f32)
#define HAVE_PKCVT 1
#endif
#endif

// two f32 -> packed bf16x2 in one dword (lo = a, hi = b)
static __device__ __forceinline__ unsigned pack2(float a, float b) {
#if HAVE_PKCVT
  auto r = __builtin_amdgcn_cvt_pk_bf16_f32(a, b);   // v_cvt_pk_bf16_f32
  return __builtin_bit_cast(unsigned, r);
#else
  unsigned short lo = __builtin_bit_cast(unsigned short, (__bf16)a);
  unsigned short hi = __builtin_bit_cast(unsigned short, (__bf16)b);
  return (unsigned)lo | ((unsigned)hi << 16);
#endif
}
// 16 contiguous f32 -> v16bf (B-operand fragment from global), dword-packed
static __device__ __forceinline__ v16bf cvt16(const float* __restrict__ p) {
  float4 f0 = *(const float4*)(p + 0);
  float4 f1 = *(const float4*)(p + 4);
  float4 f2 = *(const float4*)(p + 8);
  float4 f3 = *(const float4*)(p + 12);
  u32x8 w;
  w[0] = pack2(f0.x, f0.y); w[1] = pack2(f0.z, f0.w);
  w[2] = pack2(f1.x, f1.y); w[3] = pack2(f1.z, f1.w);
  w[4] = pack2(f2.x, f2.y); w[5] = pack2(f2.z, f2.w);
  w[6] = pack2(f3.x, f3.y); w[7] = pack2(f3.z, f3.w);
  return __builtin_bit_cast(v16bf, w);
}
// 16 contiguous LDS bf16 -> v16bf (A-operand fragment)
static __device__ __forceinline__ v16bf lds16(const unsigned short* p) {
  u32x4 lo = *(const u32x4*)(p);
  u32x4 hi = *(const u32x4*)(p + 8);
  u32x8 w = __builtin_shufflevector(lo, hi, 0, 1, 2, 3, 4, 5, 6, 7);
  return __builtin_bit_cast(v16bf, w);
}
// B-operand fragment: 8 halves at p, 8 halves at p+16
static __device__ __forceinline__ v16bf lds8x2(const unsigned short* p) {
  u32x4 lo = *(const u32x4*)(p);
  u32x4 hi = *(const u32x4*)(p + 16);
  u32x8 w = __builtin_shufflevector(lo, hi, 0, 1, 2, 3, 4, 5, 6, 7);
  return __builtin_bit_cast(v16bf, w);
}

#if HAVE_TDM
// TDM: async DMA of one 32(f32)x128-row R tile (row stride Mc) into LDS,
// with hardware row padding 32dw + 4dw == SR(36) stride. ISA D# per §8.3-8.6.
static __device__ __forceinline__ void tdm_load_R(const float* gsrc,
                                                  const float* lds_dst) {
  constexpr unsigned TD = 1u << 20;  // generous tensor dims (all accesses in-bounds)
  unsigned long long ga = (unsigned long long)(uintptr_t)gsrc;
  u32x4 g0;
  g0[0] = 1u;                                   // count=1, user descriptor
  g0[1] = (unsigned)(uintptr_t)lds_dst;         // low 32 bits = LDS byte offset
  g0[2] = (unsigned)ga;                         // global_addr[31:0]
  g0[3] = (unsigned)((ga >> 32) & 0x01FFFFFFull) | 0x80000000u;  // addr[56:32], type=2
  i32x8 g1;
  g1[0] = (int)((2u << 16)      // data_size = 4B
              | (1u << 20)      // pad_enable
              | (4u << 22)      // pad_interval: 2^(4+1)=32 dwords
              | (3u << 25));    // pad_amount: 3+1=4 dwords
  g1[1] = (int)((TD & 0xFFFFu) << 16);                            // tensor_dim0 lo
  g1[2] = (int)(((TD >> 16) & 0xFFFFu) | ((TD & 0xFFFFu) << 16)); // dim0 hi | dim1 lo
  g1[3] = (int)(((TD >> 16) & 0xFFFFu) | (32u << 16));            // dim1 hi | tile_dim0=32
  g1[4] = 128;                                                    // tile_dim1=128 rows
  g1[5] = Mc;                                                     // tensor_dim0_stride lo
  g1[6] = 0;
  g1[7] = 0;
  i32x4 z4 = (i32x4){0, 0, 0, 0};
#if __clang_major__ >= 23
  i32x8 z8 = (i32x8){0, 0, 0, 0, 0, 0, 0, 0};
  __builtin_amdgcn_tensor_load_to_lds(g0, g1, z4, z4, z8, 0);
#else
  __builtin_amdgcn_tensor_load_to_lds(g0, g1, z4, z4, 0);
#endif
}
#endif

__global__ __launch_bounds__(256) void ffnmoe_wmma(
    const float* __restrict__ Q, const float* __restrict__ K,
    const float* __restrict__ U, const float* __restrict__ V,
    const float* __restrict__ R, float* __restrict__ O) {
  __shared__ __align__(16) unsigned short sK[32 * SK];
  __shared__ __align__(16) unsigned short sU[32 * SK];
  __shared__ __align__(16) unsigned short sVt[128 * SV];
  __shared__ __align__(16) float sR2[2][128 * SR];   // double-buffered R tile

  const int tid  = threadIdx.x;
  const int lane = tid & 31;
  const int wave = tid >> 5;        // 0..7
  const int ln   = lane & 15;       // column within 16
  const int hl   = lane >> 4;       // 0: lanes 0-15, 1: lanes 16-31

  const int bh = blockIdx.x >> 4;   // 0..15 head
  const int q0 = (blockIdx.x & 15) * BQ;

  const float* Qp = Q + ((size_t)bh * Nc + q0) * Dc;
  const float* Kp = K + (size_t)bh * Mc * Dc;
  const float* Up = U + (size_t)bh * Mc * Dc;
  const float* Vp = V + (size_t)bh * Mc * Dc;
  const float* Rp = R + ((size_t)bh * Nc + q0) * (size_t)Mc;

  // ---- Q as resident B-operand: lane column ln = q-row, K dim = D ----
  v16bf Qb[4];
  {
    const float* qr = Qp + (size_t)(16 * wave + ln) * Dc;
#pragma unroll
    for (int c = 0; c < 4; ++c)
      Qb[c] = cvt16(qr + 32 * c + 16 * hl);
  }

  v8f Oc[8];
#pragma unroll
  for (int i = 0; i < 8; ++i) Oc[i] = (v8f){0.f,0.f,0.f,0.f,0.f,0.f,0.f,0.f};

  // Pointer-swap double buffer (no parity index -> no forced loop unroll).
  float* rcur = &sR2[0][0];
  float* rnxt = &sR2[1][0];

#if HAVE_TDM
  if (wave == 0) tdm_load_R(Rp, rcur);   // R tile 0 in flight
#endif

#pragma unroll 1
  for (int t = 0; t < NT; ++t) {
    const int m0 = t * TM;
    const bool more = (t + 1 < NT);
    // ---------- cooperative staging of K/U/V tile t (f32 -> bf16) ----------
#pragma unroll
    for (int i = 0; i < 4; ++i) {
      int idx = tid + 256 * i;       // 0..1023 float4 over 32x128 tile
      int row = idx >> 5;            // m-local 0..31
      int c4  = idx & 31;            // float4 column
      const size_t goff = (size_t)(m0 + row) * Dc + c4 * 4;
      float4 kf = *(const float4*)(Kp + goff);
      float4 uf = *(const float4*)(Up + goff);
      float4 vf = *(const float4*)(Vp + goff);
      if (more) {                    // warm L2 for next tile
        __builtin_prefetch(Kp + goff + TM * Dc, 0, 0);
        __builtin_prefetch(Up + goff + TM * Dc, 0, 0);
        __builtin_prefetch(Vp + goff + TM * Dc, 0, 0);
      }
      unsigned* kd = (unsigned*)(sK + row * SK + c4 * 4);
      kd[0] = pack2(kf.x, kf.y); kd[1] = pack2(kf.z, kf.w);
      unsigned* ud = (unsigned*)(sU + row * SK + c4 * 4);
      ud[0] = pack2(uf.x, uf.y); ud[1] = pack2(uf.z, uf.w);
      int d0 = c4 * 4;               // V stored transposed: [d][m]
      sVt[(d0 + 0) * SV + row] = (unsigned short)pack2(vf.x, 0.f);
      sVt[(d0 + 1) * SV + row] = (unsigned short)pack2(vf.y, 0.f);
      sVt[(d0 + 2) * SV + row] = (unsigned short)pack2(vf.z, 0.f);
      sVt[(d0 + 3) * SV + row] = (unsigned short)pack2(vf.w, 0.f);
    }
#if HAVE_TDM
    // Kick TDM for R tile t+1 into the alternate buffer; ensure tile t landed.
    if (more) {
      if (wave == 0) {
        tdm_load_R(Rp + (m0 + TM), rnxt);
        __builtin_amdgcn_s_wait_tensorcnt(1);   // in-order: tile t complete
      }
    } else if (wave == 0) {
      __builtin_amdgcn_s_wait_tensorcnt(0);
    }
#else
    // Fallback: cooperative f32 staging of R tile t.
#pragma unroll
    for (int i = 0; i < 4; ++i) {
      int idx = tid + 256 * i;       // 0..1023 float4 over 128x32 R tile
      int row = idx >> 3;            // q-local 0..127
      int c4  = idx & 7;
      const float* rp = Rp + (size_t)row * Mc + m0 + c4 * 4;
      *(float4*)(rcur + row * SR + c4 * 4) = *(const float4*)rp;
      if (more) __builtin_prefetch(rp + TM, 0, 0);
    }
#endif
    __syncthreads();

    // ---------- compute: S^T = K*Q^T, T^T = U*Q^T, gate, O += G*V ----------
    unsigned gpk[8];                  // packed A-operand for O-GEMM
#pragma unroll
    for (int h2 = 0; h2 < 2; ++h2) {
      const unsigned short* krow = sK + (16 * h2 + ln) * SK + 16 * hl;
      const unsigned short* urow = sU + (16 * h2 + ln) * SK + 16 * hl;
      // Batch-preload all 8 A-fragments so ds_loads run ahead of the WMMA chain.
      v16bf Ka[4], Ua[4];
#pragma unroll
      for (int c = 0; c < 4; ++c) {
        Ka[c] = lds16(krow + 32 * c);
        Ua[c] = lds16(urow + 32 * c);
      }
      // R gate values (needed only after the chain) - issue loads early.
      const float* rrow = rcur + (16 * wave + ln) * SR + 16 * h2 + 8 * hl;
      float4 ra = *(const float4*)(rrow);
      float4 rb = *(const float4*)(rrow + 4);
      v8f St = (v8f){0.f,0.f,0.f,0.f,0.f,0.f,0.f,0.f};
      v8f Tt = (v8f){0.f,0.f,0.f,0.f,0.f,0.f,0.f,0.f};
#pragma unroll
      for (int c = 0; c < 4; ++c) {   // chain K-dim: D = 128 = 4 x 32
        St = __builtin_amdgcn_wmma_f32_16x16x32_bf16(
            false, Ka[c], false, Qb[c], (short)0, St, false, false);
        Tt = __builtin_amdgcn_wmma_f32_16x16x32_bf16(
            false, Ua[c], false, Qb[c], (short)0, Tt, false, false);
      }
      // Gate: G[m][q] = relu(S^T)*T^T*R ; C layout: lane = q, vgpr v -> m = v+8*hl
      float rv[8] = {ra.x, ra.y, ra.z, ra.w, rb.x, rb.y, rb.z, rb.w};
#pragma unroll
      for (int v2 = 0; v2 < 4; ++v2) {
        float g0 = __builtin_fmaxf(St[2 * v2], 0.f)     * Tt[2 * v2]     * rv[2 * v2];
        float g1 = __builtin_fmaxf(St[2 * v2 + 1], 0.f) * Tt[2 * v2 + 1] * rv[2 * v2 + 1];
        gpk[v2 + 4 * h2] = pack2(g0, g1);   // lands exactly in A-operand layout
      }
    }
    u32x8 gw;
#pragma unroll
    for (int i = 0; i < 8; ++i) gw[i] = gpk[i];
    v16bf Ga = __builtin_bit_cast(v16bf, gw);
    // O accumulation: Oc[dc] (16 q x 16 d) += Ga (16 q x 32 m) * V (32 m x 16 d)
    // Depth-1 pipeline on the Vb fragments.
    v16bf Vb = lds8x2(sVt + (16 * 0 + ln) * SV + 8 * hl);
#pragma unroll
    for (int dc = 0; dc < 8; ++dc) {
      v16bf Vn;
      if (dc < 7) Vn = lds8x2(sVt + (16 * (dc + 1) + ln) * SV + 8 * hl);
      Oc[dc] = __builtin_amdgcn_wmma_f32_16x16x32_bf16(
          false, Ga, false, Vb, (short)0, Oc[dc], false, false);
      if (dc < 7) Vb = Vn;
    }
    __syncthreads();
    // Swap R buffers.
    float* tmp = rcur; rcur = rnxt; rnxt = tmp;
  }

  // ---------- store O (f32): lane = d col, vgpr v -> q row ----------
  float* orow = O + ((size_t)bh * Nc + q0 + 16 * wave + 8 * hl) * Dc + ln;
#pragma unroll
  for (int v = 0; v < 8; ++v) {
    float* po = orow + (size_t)v * Dc;
#pragma unroll
    for (int dc = 0; dc < 8; ++dc) po[16 * dc] = Oc[dc][v];
  }
}

extern "C" void kernel_launch(void* const* d_in, const int* in_sizes, int n_in,
                              void* d_out, int out_size, void* d_ws, size_t ws_size,
                              hipStream_t stream) {
  (void)in_sizes; (void)n_in; (void)out_size; (void)d_ws; (void)ws_size;
  const float* Q = (const float*)d_in[0];
  const float* K = (const float*)d_in[1];
  const float* U = (const float*)d_in[2];
  const float* V = (const float*)d_in[3];
  const float* R = (const float*)d_in[4];
  float* O = (float*)d_out;
  dim3 grid(BH * (Nc / BQ));   // 16 heads * 16 row tiles = 256 workgroups
  ffnmoe_wmma<<<grid, 256, 0, stream>>>(Q, K, U, V, R, O);
}
